// visionTransformerLayer_42545946034610
// MI455X (gfx1250) — compile-verified, matrix operand order checked by reading, not run on hardware
//
#include <hip/hip_runtime.h>
#include <math.h>

// ---------------------------------------------------------------------------
// ViT-Base forward for MI455X (gfx1250): bf16 WMMA + TDM async tile staging.
// ---------------------------------------------------------------------------

typedef __bf16 bf16_t;
typedef __attribute__((ext_vector_type(8)))  __bf16 bf16x8;
typedef __attribute__((ext_vector_type(16))) __bf16 v16bf;
typedef __attribute__((ext_vector_type(8)))  float  v8f;
typedef unsigned int u32x4 __attribute__((ext_vector_type(4)));
typedef int          i32x4 __attribute__((ext_vector_type(4)));
typedef int          i32x8 __attribute__((ext_vector_type(8)));

#if defined(__HIP_DEVICE_COMPILE__) && __has_builtin(__builtin_amdgcn_tensor_load_to_lds) && \
    __has_builtin(__builtin_amdgcn_s_wait_tensorcnt)
#define HAVE_TDM 1
#else
#define HAVE_TDM 0
#endif

#define WMMA_BF16(a, b, c) \
  __builtin_amdgcn_wmma_f32_16x16x32_bf16(false, (a), false, (b), (short)0, (c), false, false)

// Load one 16x32-bf16 A/B fragment half-row: elements 0..7 at p, 8..15 at p+16
// (ISA 7.12.2: lanes 0-15 K=0..7/16..23, lanes 16-31 K=8..15/24..31; caller
//  bakes (lane>>4)*8 into p).
__device__ __forceinline__ v16bf ldfrag(const bf16_t* p) {
  bf16x8 lo = *reinterpret_cast<const bf16x8*>(p);
  bf16x8 hi = *reinterpret_cast<const bf16x8*>(p + 16);
  v16bf r;
#pragma unroll
  for (int i = 0; i < 8; ++i) { r[i] = lo[i]; r[i + 8] = hi[i]; }
  return r;
}

__device__ __forceinline__ float gelu_f(float z) {
  float z3 = z * z * z;
  return 0.5f * z * (1.0f + tanhf(0.7978845608028654f * (z + 0.044715f * z3)));
}

// ------------------------------ constants ----------------------------------
#define BATCH 16
#define SEQ   197
#define DIM   768
#define NHEAD 12
#define DHEAD 64
#define NLAY  12
#define FFD   3072
#define NPAT  196
#define TOK   (BATCH * SEQ)   /* 3152 */
#define TOKP  (BATCH * NPAT)  /* 3136 */
#define KPAD  224             /* padded key length for P@V */

// ------------------------------ TDM helper ---------------------------------
#if HAVE_TDM
// 2D bf16 tile load: tile_dim0=32 elems (64B rows) x tile_dim1=128 rows,
// LDS padded to 40-elem rows via pad_interval=16 DW (val 3), pad_amount=4 DW
// (val 3).  tensor_dim1 < 128 => TDM zero-fills OOB rows (M-tail guard).
__device__ __forceinline__ void tdm_load_tile(unsigned lds_addr, const void* gaddr,
                                              unsigned tensor_d0, unsigned tensor_d1,
                                              unsigned long long stride0_elems) {
  unsigned long long ga = (unsigned long long)(size_t)gaddr;
  u32x4 g0;
  g0[0] = 1u;                                             // count=1 (user D#)
  g0[1] = lds_addr;                                       // lds_addr [63:32]
  g0[2] = (unsigned)(ga & 0xffffffffu);                   // global_addr [95:64]
  g0[3] = (unsigned)((ga >> 32) & 0x01ffffffu) | (2u << 30);  // addr[56:32] | type=2
  i32x8 g1;
  g1[0] = (int)((1u << 16) | (1u << 20) | (3u << 22) | (3u << 25));
  //            data_size=2B  pad_en      pad_int=16DW  pad_amt=4DW
  g1[1] = (int)((tensor_d0 & 0xffffu) << 16);             // tensor_dim0[15:0]
  g1[2] = (int)((tensor_d0 >> 16) | ((tensor_d1 & 0xffffu) << 16));
  g1[3] = (int)((tensor_d1 >> 16) | (32u << 16));         // tile_dim0 = 32 elems
  g1[4] = (int)128;                                       // tile_dim1 = 128 rows
  g1[5] = (int)(unsigned)(stride0_elems & 0xffffffffull); // tensor_dim0_stride lo
  g1[6] = (int)(unsigned)((stride0_elems >> 32) & 0xffffull);
  g1[7] = 0;
  i32x4 z4;
  z4[0] = 0; z4[1] = 0; z4[2] = 0; z4[3] = 0;
#if __clang_major__ >= 23
  i32x8 z8;
#pragma unroll
  for (int i = 0; i < 8; ++i) z8[i] = 0;
  __builtin_amdgcn_tensor_load_to_lds(g0, g1, z4, z4, z8, 0);
#else
  __builtin_amdgcn_tensor_load_to_lds(g0, g1, z4, z4, 0);
#endif
}
#endif

// ------------------------------ GEMM ---------------------------------------
enum { EPI_F32_BIAS = 0, EPI_F32_RESID = 1, EPI_BF16_QK = 2, EPI_BF16_VT = 3, EPI_BF16_GELU = 4 };

#define GBM 128
#define GBN 128
#define GBK 32
#define GLDK 40  // padded LDS K-stride (bf16): 80B rows, 16B aligned

// C = A[M,K](bf16,row-major) * Wt[N,K](bf16,row-major == W^T) with f32 acc.
// grid = (N/128, ceil(M/128)), block = 256 (8 waves, 4x2 wave grid, 2x4 tiles/wave)
// Double-buffered: TDM streams tile ks+1 into LDS while waves WMMA tile ks.
template <int EPI>
__global__ __launch_bounds__(256) void gemm_wmma(
    const bf16_t* __restrict__ A, const bf16_t* __restrict__ Bt,
    const float* __restrict__ bias, const float* __restrict__ resid,
    float* __restrict__ outf, bf16_t* __restrict__ outb, int M, int N, int K) {
  __shared__ __attribute__((aligned(16))) bf16_t sA[2][GBM * GLDK];
  __shared__ __attribute__((aligned(16))) bf16_t sB[2][GBN * GLDK];

  const int tid = threadIdx.x;
  const int row0 = blockIdx.y * GBM, col0 = blockIdx.x * GBN;
  const int w = tid >> 5, lane = tid & 31;
  const int wm = w & 3, wn = w >> 2;
  const int l16 = lane & 15, lhalf = lane >> 4;

  v8f acc[2][4];
#pragma unroll
  for (int i = 0; i < 2; ++i)
#pragma unroll
    for (int j = 0; j < 4; ++j)
#pragma unroll
      for (int e = 0; e < 8; ++e) acc[i][j][e] = 0.f;

  const int KS = K / GBK;
  const int col = lhalf * 8;

#if HAVE_TDM
  // ---- TDM double-buffered pipeline ----
  if (tid < 32) {
    tdm_load_tile((unsigned)(size_t)(void*)&sA[0][0], A + (size_t)row0 * K,
                  (unsigned)K, (unsigned)(M - row0), (unsigned long long)K);
    tdm_load_tile((unsigned)(size_t)(void*)&sB[0][0], Bt + (size_t)col0 * K,
                  (unsigned)K, (unsigned)(N - col0), (unsigned long long)K);
    __builtin_amdgcn_s_wait_tensorcnt(0);
  }
  __syncthreads();
  for (int ks = 0; ks < KS; ++ks) {
    const int cur = ks & 1;
    if (tid < 32 && ks + 1 < KS) {
      const int k0 = (ks + 1) * GBK;
      tdm_load_tile((unsigned)(size_t)(void*)&sA[cur ^ 1][0],
                    A + (size_t)row0 * K + k0, (unsigned)(K - k0),
                    (unsigned)(M - row0), (unsigned long long)K);
      tdm_load_tile((unsigned)(size_t)(void*)&sB[cur ^ 1][0],
                    Bt + (size_t)col0 * K + k0, (unsigned)(K - k0),
                    (unsigned)(N - col0), (unsigned long long)K);
    }
    {
      v16bf af[2], bfr[4];
#pragma unroll
      for (int i = 0; i < 2; ++i)
        af[i] = ldfrag(&sA[cur][(wm * 32 + i * 16 + l16) * GLDK + col]);
#pragma unroll
      for (int j = 0; j < 4; ++j)
        bfr[j] = ldfrag(&sB[cur][(wn * 64 + j * 16 + l16) * GLDK + col]);
#pragma unroll
      for (int i = 0; i < 2; ++i)
#pragma unroll
        for (int j = 0; j < 4; ++j) acc[i][j] = WMMA_BF16(af[i], bfr[j], acc[i][j]);
    }
    __syncthreads();  // all waves done reading buf[cur]
    if (tid < 32 && ks + 1 < KS) __builtin_amdgcn_s_wait_tensorcnt(0);
    __syncthreads();  // publish buf[cur^1]
  }
#else
  // ---- fallback: vector-load double buffer ----
  auto stage = [&](int ks, int buf) {
    const int k0 = ks * GBK;
#pragma unroll
    for (int i = 0; i < 2; ++i) {
      int vi = tid + i * 256;
      int r = vi >> 2, cv = vi & 3;
      int gr = row0 + r;
      bf16x8 v;
#pragma unroll
      for (int t = 0; t < 8; ++t) v[t] = (bf16_t)0.f;
      if (gr < M) v = *reinterpret_cast<const bf16x8*>(A + (size_t)gr * K + k0 + cv * 8);
      *reinterpret_cast<bf16x8*>(&sA[buf][r * GLDK + cv * 8]) = v;
    }
#pragma unroll
    for (int i = 0; i < 2; ++i) {
      int vi = tid + i * 256;
      int r = vi >> 2, cv = vi & 3;
      bf16x8 v = *reinterpret_cast<const bf16x8*>(Bt + (size_t)(col0 + r) * K + k0 + cv * 8);
      *reinterpret_cast<bf16x8*>(&sB[buf][r * GLDK + cv * 8]) = v;
    }
  };
  stage(0, 0);
  __syncthreads();
  for (int ks = 0; ks < KS; ++ks) {
    const int cur = ks & 1;
    if (ks + 1 < KS) stage(ks + 1, cur ^ 1);
    {
      v16bf af[2], bfr[4];
#pragma unroll
      for (int i = 0; i < 2; ++i)
        af[i] = ldfrag(&sA[cur][(wm * 32 + i * 16 + l16) * GLDK + col]);
#pragma unroll
      for (int j = 0; j < 4; ++j)
        bfr[j] = ldfrag(&sB[cur][(wn * 64 + j * 16 + l16) * GLDK + col]);
#pragma unroll
      for (int i = 0; i < 2; ++i)
#pragma unroll
        for (int j = 0; j < 4; ++j) acc[i][j] = WMMA_BF16(af[i], bfr[j], acc[i][j]);
    }
    __syncthreads();
  }
#endif

  // epilogue (C layout: lane n = lane&15, vgpr e -> m = e + (lane>>4)*8)
#pragma unroll
  for (int i = 0; i < 2; ++i) {
#pragma unroll
    for (int j = 0; j < 4; ++j) {
      const int n = col0 + wn * 64 + j * 16 + l16;
      const float bn = bias ? bias[n] : 0.f;
#pragma unroll
      for (int e = 0; e < 8; ++e) {
        const int m = row0 + wm * 32 + i * 16 + lhalf * 8 + e;
        if (m >= M) continue;
        float val = acc[i][j][e] + bn;
        if (EPI == EPI_F32_BIAS) {
          outf[(size_t)m * N + n] = val;
        } else if (EPI == EPI_F32_RESID) {
          outf[(size_t)m * N + n] = resid[(size_t)m * N + n] + val;
        } else if (EPI == EPI_BF16_QK) {
          int b = m / SEQ, s = m - b * SEQ;
          int hh = n >> 6, dh = n & 63;
          outb[(((size_t)b * NHEAD + hh) * SEQ + s) * DHEAD + dh] = (bf16_t)val;
        } else if (EPI == EPI_BF16_VT) {
          int b = m / SEQ, s = m - b * SEQ;
          int hh = n >> 6, dh = n & 63;
          outb[(((size_t)b * NHEAD + hh) * DHEAD + dh) * KPAD + s] = (bf16_t)val;
        } else {  // EPI_BF16_GELU
          outb[(size_t)m * N + n] = (bf16_t)gelu_f(val);
        }
      }
    }
  }
}

// -------------------------- weight convert+transpose ------------------------
// W[K,N] f32 -> Wt[N,K] bf16. grid(N/32, K/32), block 256.
__global__ __launch_bounds__(256) void transpose_wt(const float* __restrict__ W,
                                                    bf16_t* __restrict__ Wt, int K, int N) {
  __shared__ float tile[32][33];
  const int n0 = blockIdx.x * 32, k0 = blockIdx.y * 32;
  const int tx = threadIdx.x & 31, ty = threadIdx.x >> 5;  // 32 x 8
#pragma unroll
  for (int r = ty; r < 32; r += 8) tile[r][tx] = W[(size_t)(k0 + r) * N + n0 + tx];
  __syncthreads();
#pragma unroll
  for (int r = ty; r < 32; r += 8)
    Wt[(size_t)(n0 + r) * K + k0 + tx] = (bf16_t)tile[tx][r];
}

// ------------------------------ layernorm ----------------------------------
template <bool OUT_BF16>
__global__ __launch_bounds__(256) void layernorm_k(const float* __restrict__ in,
                                                   const float* __restrict__ g,
                                                   const float* __restrict__ bta,
                                                   bf16_t* __restrict__ outb,
                                                   float* __restrict__ outf, float eps) {
  const int row = blockIdx.x, tid = threadIdx.x;
  const float* p = in + (size_t)row * DIM;
  float v0 = p[tid], v1 = p[tid + 256], v2 = p[tid + 512];
  __shared__ float r1[256], r2[256];
  r1[tid] = v0 + v1 + v2;
  r2[tid] = v0 * v0 + v1 * v1 + v2 * v2;
  __syncthreads();
  for (int o = 128; o > 0; o >>= 1) {
    if (tid < o) { r1[tid] += r1[tid + o]; r2[tid] += r2[tid + o]; }
    __syncthreads();
  }
  const float mean = r1[0] * (1.f / DIM);
  const float var = r2[0] * (1.f / DIM) - mean * mean;
  const float rstd = rsqrtf(var + eps);
  float vv[3] = {v0, v1, v2};
#pragma unroll
  for (int t = 0; t < 3; ++t) {
    int c = tid + t * 256;
    float o = (vv[t] - mean) * rstd * g[c] + bta[c];
    if (OUT_BF16) outb[(size_t)row * DIM + c] = (bf16_t)o;
    else          outf[(size_t)row * DIM + c] = o;
  }
}

// ------------------------------ attention ----------------------------------
// one workgroup per (b,h). q,k: [BH,197,64] bf16; vt: [BH,64,224] bf16 (pad zeroed)
// att out: [BH,197,64] bf16 (== heads-interleaved [B,S,D] raw reshape)
__global__ __launch_bounds__(256) void attention_k(const bf16_t* __restrict__ q,
                                                   const bf16_t* __restrict__ k,
                                                   const bf16_t* __restrict__ vt,
                                                   bf16_t* __restrict__ att) {
  const int bh = blockIdx.x;
  const int tid = threadIdx.x, w = tid >> 5, lane = tid & 31;
  const int l16 = lane & 15, lhalf = lane >> 4;
  __shared__ float sS[32 * KPAD];
  __shared__ __attribute__((aligned(16))) bf16_t sP[32 * 232];
  const bf16_t* qb = q + (size_t)bh * SEQ * DHEAD;
  const bf16_t* kb = k + (size_t)bh * SEQ * DHEAD;
  const bf16_t* vb = vt + (size_t)bh * DHEAD * KPAD;
  bf16_t* ab = att + (size_t)bh * SEQ * DHEAD;

  for (int m0 = 0; m0 < 208; m0 += 32) {
    const int tiles_m = ((208 - m0) >> 4) < 2 ? ((208 - m0) >> 4) : 2;
    // ---- scores = (q @ k^T) * 1/sqrt(64); Bt operand is k itself ----
    const int ntile = tiles_m * 13;
    for (int idx = w; idx < ntile; idx += 8) {
      const int tm = idx / 13, tn = idx - tm * 13;
      v8f a;
#pragma unroll
      for (int e = 0; e < 8; ++e) a[e] = 0.f;
#pragma unroll
      for (int kk = 0; kk < 64; kk += 32) {
        const int col = kk + lhalf * 8;
        v16bf af = ldfrag(qb + ((size_t)(m0 + tm * 16 + l16)) * DHEAD + col);
        v16bf bf = ldfrag(kb + ((size_t)(tn * 16 + l16)) * DHEAD + col);
        a = WMMA_BF16(af, bf, a);
      }
#pragma unroll
      for (int e = 0; e < 8; ++e)
        sS[(tm * 16 + lhalf * 8 + e) * KPAD + tn * 16 + l16] = a[e] * 0.125f;
    }
    __syncthreads();
    // ---- softmax over valid keys [0,197); P=0 in pad cols [197,224) ----
    const int rows = tiles_m * 16;
    if (tid < rows) {
      float* srow = &sS[tid * KPAD];
      float mx = -1e30f;
      for (int n = 0; n < SEQ; ++n) mx = fmaxf(mx, srow[n]);
      float sum = 0.f;
      for (int n = 0; n < SEQ; ++n) { float e = __expf(srow[n] - mx); srow[n] = e; sum += e; }
      const float inv = 1.f / sum;
      bf16_t* prow = &sP[tid * 232];
      for (int n = 0; n < SEQ; ++n) prow[n] = (bf16_t)(srow[n] * inv);
      for (int n = SEQ; n < KPAD; ++n) prow[n] = (bf16_t)0.f;
    }
    __syncthreads();
    // ---- att = P @ V ; Bt operand is v^T rows (dh-major) ----
    for (int idx = w; idx < tiles_m * 4; idx += 8) {
      const int tm = idx >> 2, tn = idx & 3;
      v8f a;
#pragma unroll
      for (int e = 0; e < 8; ++e) a[e] = 0.f;
#pragma unroll
      for (int ks = 0; ks < KPAD; ks += 32) {
        const int col = ks + lhalf * 8;
        v16bf af = ldfrag(&sP[(tm * 16 + l16) * 232 + col]);
        v16bf bf = ldfrag(vb + ((size_t)(tn * 16 + l16)) * KPAD + col);
        a = WMMA_BF16(af, bf, a);
      }
#pragma unroll
      for (int e = 0; e < 8; ++e) {
        const int s = m0 + tm * 16 + lhalf * 8 + e;
        if (s < SEQ) ab[(size_t)s * DHEAD + tn * 16 + l16] = (bf16_t)a[e];
      }
    }
    __syncthreads();
  }
}

// ---------------------------- small kernels --------------------------------
__global__ void patch_extract_k(const float* __restrict__ x, bf16_t* __restrict__ patches) {
  int idx = blockIdx.x * 256 + threadIdx.x;
  if (idx >= TOKP * DIM) return;
  int row = idx / DIM, c = idx - row * DIM;
  int b = row / NPAT, p = row - b * NPAT;
  int gy = p / 14, gx = p - gy * 14;
  int ch = c % 3, q2 = c / 3;
  int px = q2 & 15, py = q2 >> 4;
  size_t xi = (((size_t)b * 224 + gy * 16 + py) * 224 + gx * 16 + px) * 3 + ch;
  patches[idx] = (bf16_t)x[xi];
}

__global__ void assemble_k(const float* __restrict__ hp, const float* __restrict__ cls,
                           const float* __restrict__ pos, float* __restrict__ h) {
  int idx = blockIdx.x * 256 + threadIdx.x;
  if (idx >= TOK * DIM) return;
  int t = idx / DIM, d = idx - t * DIM;
  int b = t / SEQ, s = t - b * SEQ;
  float v = (s == 0) ? cls[d] : hp[((size_t)b * NPAT + (s - 1)) * DIM + d];
  h[idx] = v + pos[(size_t)s * DIM + d];
}

__global__ void zfill_k(bf16_t* __restrict__ p, int n) {
  int i = blockIdx.x * 256 + threadIdx.x;
  if (i < n) p[i] = (bf16_t)0.f;
}

// ------------------------------ launcher -----------------------------------
extern "C" void kernel_launch(void* const* d_in, const int* in_sizes, int n_in,
                              void* d_out, int out_size, void* d_ws, size_t ws_size,
                              hipStream_t stream) {
  (void)in_sizes; (void)n_in; (void)out_size; (void)ws_size;
  const float* x    = (const float*)d_in[0];
  const float* cls  = (const float*)d_in[1];
  const float* pos  = (const float*)d_in[2];
  const float* Wp   = (const float*)d_in[3];
  const float* bp   = (const float*)d_in[4];
  const float* Wq   = (const float*)d_in[5];
  const float* bq   = (const float*)d_in[6];
  const float* Wk   = (const float*)d_in[7];
  const float* bk   = (const float*)d_in[8];
  const float* Wv   = (const float*)d_in[9];
  const float* bv   = (const float*)d_in[10];
  const float* Wo   = (const float*)d_in[11];
  const float* bo   = (const float*)d_in[12];
  const float* ln1g = (const float*)d_in[13];
  const float* ln1b = (const float*)d_in[14];
  const float* ln2g = (const float*)d_in[15];
  const float* ln2b = (const float*)d_in[16];
  const float* W1   = (const float*)d_in[17];
  const float* b1   = (const float*)d_in[18];
  const float* W2   = (const float*)d_in[19];
  const float* b2   = (const float*)d_in[20];
  const float* lnfg = (const float*)d_in[21];
  const float* lnfb = (const float*)d_in[22];

  // ---- workspace carve-up ----
  char* ws = (char*)d_ws;
  size_t off = 0;
  auto alloc = [&](size_t bytes) -> void* {
    void* p = ws + off;
    off = (off + bytes + 255) & ~(size_t)255;
    return p;
  };
  bf16_t* patches = (bf16_t*)alloc((size_t)TOKP * DIM * 2);
  float*  hp      = (float*) alloc((size_t)TOKP * DIM * 4);
  float*  h       = (float*) alloc((size_t)TOK * DIM * 4);
  float*  h1      = (float*) alloc((size_t)TOK * DIM * 4);
  bf16_t* y       = (bf16_t*)alloc((size_t)TOK * DIM * 2);
  bf16_t* qb      = (bf16_t*)alloc((size_t)TOK * DIM * 2 + 4096);  // pad: attn reads s up to 207
  bf16_t* kb      = (bf16_t*)alloc((size_t)TOK * DIM * 2 + 4096);
  bf16_t* vtb     = (bf16_t*)alloc((size_t)BATCH * NHEAD * DHEAD * KPAD * 2);
  bf16_t* attb    = (bf16_t*)alloc((size_t)TOK * DIM * 2);
  bf16_t* zb      = (bf16_t*)alloc((size_t)TOK * FFD * 2);
  bf16_t* wqt     = (bf16_t*)alloc((size_t)DIM * DIM * 2);
  bf16_t* wkt     = (bf16_t*)alloc((size_t)DIM * DIM * 2);
  bf16_t* wvt     = (bf16_t*)alloc((size_t)DIM * DIM * 2);
  bf16_t* wot     = (bf16_t*)alloc((size_t)DIM * DIM * 2);
  bf16_t* w1t     = (bf16_t*)alloc((size_t)FFD * DIM * 2);
  bf16_t* w2t     = (bf16_t*)alloc((size_t)DIM * FFD * 2);
  bf16_t* wpt     = (bf16_t*)alloc((size_t)DIM * DIM * 2);

  const dim3 blk(256);
  const dim3 gD (DIM / 128, (TOK + 127) / 128);   // N=768 GEMMs over tokens
  const dim3 gF (FFD / 128, (TOK + 127) / 128);   // N=3072 GEMM
  const dim3 gP (DIM / 128, (TOKP + 127) / 128);  // patch GEMM
  const dim3 gTd(DIM / 32, DIM / 32);             // 768x768 transpose
  const dim3 gT1(FFD / 32, DIM / 32);             // W1: K=768,N=3072
  const dim3 gT2(DIM / 32, FFD / 32);             // W2: K=3072,N=768

  // zero V^T pad region (whole buffer) once per call
  {
    int nvt = BATCH * NHEAD * DHEAD * KPAD;
    zfill_k<<<(nvt + 255) / 256, blk, 0, stream>>>(vtb, nvt);
  }

  // ---- patch embedding ----
  patch_extract_k<<<((TOKP * DIM) + 255) / 256, blk, 0, stream>>>(x, patches);
  transpose_wt<<<gTd, blk, 0, stream>>>(Wp, wpt, DIM, DIM);
  gemm_wmma<EPI_F32_BIAS><<<gP, blk, 0, stream>>>(patches, wpt, bp, nullptr, hp, nullptr,
                                                  TOKP, DIM, DIM);
  assemble_k<<<((TOK * DIM) + 255) / 256, blk, 0, stream>>>(hp, cls, pos, h);

  // ---- transformer blocks ----
  for (int l = 0; l < NLAY; ++l) {
    const size_t wD = (size_t)l * DIM * DIM;
    const size_t wF = (size_t)l * DIM * FFD;
    transpose_wt<<<gTd, blk, 0, stream>>>(Wq + wD, wqt, DIM, DIM);
    transpose_wt<<<gTd, blk, 0, stream>>>(Wk + wD, wkt, DIM, DIM);
    transpose_wt<<<gTd, blk, 0, stream>>>(Wv + wD, wvt, DIM, DIM);
    transpose_wt<<<gTd, blk, 0, stream>>>(Wo + wD, wot, DIM, DIM);
    transpose_wt<<<gT1, blk, 0, stream>>>(W1 + wF, w1t, DIM, FFD);
    transpose_wt<<<gT2, blk, 0, stream>>>(W2 + wF, w2t, FFD, DIM);

    layernorm_k<true><<<TOK, blk, 0, stream>>>(h, ln1g + l * DIM, ln1b + l * DIM,
                                               y, nullptr, 1e-6f);
    gemm_wmma<EPI_BF16_QK><<<gD, blk, 0, stream>>>(y, wqt, bq + l * DIM, nullptr,
                                                   nullptr, qb, TOK, DIM, DIM);
    gemm_wmma<EPI_BF16_QK><<<gD, blk, 0, stream>>>(y, wkt, bk + l * DIM, nullptr,
                                                   nullptr, kb, TOK, DIM, DIM);
    gemm_wmma<EPI_BF16_VT><<<gD, blk, 0, stream>>>(y, wvt, bv + l * DIM, nullptr,
                                                   nullptr, vtb, TOK, DIM, DIM);
    attention_k<<<BATCH * NHEAD, blk, 0, stream>>>(qb, kb, vtb, attb);
    gemm_wmma<EPI_F32_RESID><<<gD, blk, 0, stream>>>(attb, wot, bo + l * DIM, h,
                                                     h1, nullptr, TOK, DIM, DIM);
    layernorm_k<true><<<TOK, blk, 0, stream>>>(h1, ln2g + l * DIM, ln2b + l * DIM,
                                               y, nullptr, 1e-6f);
    gemm_wmma<EPI_BF16_GELU><<<gF, blk, 0, stream>>>(y, w1t, b1 + l * FFD, nullptr,
                                                     nullptr, zb, TOK, FFD, DIM);
    gemm_wmma<EPI_F32_RESID><<<gD, blk, 0, stream>>>(zb, w2t, b2 + l * DIM, h1,
                                                     h, nullptr, TOK, DIM, FFD);
  }

  // ---- final LayerNorm (Keras default eps=1e-3), f32 out ----
  layernorm_k<false><<<TOK, blk, 0, stream>>>(h, lnfg, lnfb, nullptr, (float*)d_out, 1e-3f);
}
